// AttentionBlock_7378753814534
// MI455X (gfx1250) — compile-verified, matrix operand order checked by reading, not run on hardware
//
#include <hip/hip_runtime.h>

typedef __bf16 bf16;
typedef __attribute__((ext_vector_type(16))) __bf16 v16bf;
typedef __attribute__((ext_vector_type(8)))  __bf16 v8bf;
typedef __attribute__((ext_vector_type(8)))  float  v8f;

#define B_    8
#define C_    256
#define S_    1024
#define NH    8
#define DK    64
#define INNER 512
#define N3    1536
#define ATT_SCALE 0.125f   // DK^-0.5

// ---------------------------------------------------------------------------
// D = A*B + C, bf16 inputs, fp32 accumulate (v_wmma_f32_16x16x32_bf16)
__device__ __forceinline__ v8f wmma_bf16(v16bf a, v16bf b, v8f c) {
  return __builtin_amdgcn_wmma_f32_16x16x32_bf16(false, a, false, b,
                                                 (short)0, c, false, false);
}

// Load a 16x32 bf16 fragment (A layout; also Bt layout: rows = N, cols = K)
// from a row-major bf16 buffer with row stride `stride` elements.
// Per ISA 7.12.2: lane m=lane&15 holds row m; VGPRs 0..3 hold K = kb..kb+7,
// VGPRs 4..7 hold K = 16+kb..16+kb+7, where kb = 8*(lane>=16).
__device__ __forceinline__ v16bf load_frag(const bf16* base, int r0, int c0,
                                           int stride, int lane) {
  const int m  = lane & 15;
  const int kb = (lane >> 4) << 3;
  const bf16* p = base + (size_t)(r0 + m) * stride + c0 + kb;
  union { v16bf v; v8bf h[2]; } u;
  u.h[0] = *(const v8bf*)(p);
  u.h[1] = *(const v8bf*)(p + 16);
  return u.v;
}

// ---------------------------------------------------------------------------
// Kernel 1: qkv = xt @ w_qkv + b_qkv ; scatter into q/k/v [B,H,S,DK] bf16
// Tile: 128(M) x 128(N), 8 waves as 2x4, each wave 64x32, K-step 32.
__global__ __launch_bounds__(256)
void qkv_kernel(const float* __restrict__ x, const float* __restrict__ w_qkv,
                const float* __restrict__ b_qkv,
                bf16* __restrict__ qws, bf16* __restrict__ kws,
                bf16* __restrict__ vws)
{
  const int M0 = blockIdx.x * 128;
  const int N0 = blockIdx.y * 128;
  const int tid = threadIdx.x;
  const int lane = tid & 31, wv = tid >> 5;
  const int wm = wv & 1, wn = wv >> 1;

  __shared__ bf16 As[128 * 40];   // A tile  [m][k], k-stride 40 (16B aligned rows)
  __shared__ bf16 Bs[128 * 40];   // B^T tile [n][k]

  const int bIdx  = M0 >> 10;     // 128-row tile never crosses a batch (1024%128==0)
  const int sBase = M0 & 1023;

  v8f acc[4][2];
  #pragma unroll
  for (int i = 0; i < 4; ++i)
    #pragma unroll
    for (int j = 0; j < 2; ++j)
      acc[i][j] = v8f{0.f,0.f,0.f,0.f,0.f,0.f,0.f,0.f};

  for (int k0 = 0; k0 < C_; k0 += 32) {
    // Stage A: As[r][c] = xt[M0+r, k0+c] = x[bIdx, k0+c, sBase+r]
    {
      const int c  = tid >> 3;          // 0..31
      const int r0 = (tid & 7) << 4;    // 0..112
      const float* src = x + ((size_t)(bIdx * C_ + k0 + c) * S_ + sBase + r0);
      if (k0 + 32 < C_) __builtin_prefetch(src + 32 * S_, 0, 1);
      #pragma unroll
      for (int r = 0; r < 16; r += 4) {
        float4 f = *(const float4*)(src + r);
        As[(r0 + r + 0) * 40 + c] = (bf16)f.x;
        As[(r0 + r + 1) * 40 + c] = (bf16)f.y;
        As[(r0 + r + 2) * 40 + c] = (bf16)f.z;
        As[(r0 + r + 3) * 40 + c] = (bf16)f.w;
      }
    }
    // Stage B^T: Bs[n][k] = w_qkv[k0+k, N0+n]
    {
      const int r  = tid >> 3;          // k_local 0..31
      const int c0 = (tid & 7) << 4;    // n_local 0..112
      const float* src = w_qkv + ((size_t)(k0 + r) * N3 + N0 + c0);
      if (k0 + 32 < C_) __builtin_prefetch(src + 32 * N3, 0, 1);
      #pragma unroll
      for (int n = 0; n < 16; n += 4) {
        float4 f = *(const float4*)(src + n);
        Bs[(c0 + n + 0) * 40 + r] = (bf16)f.x;
        Bs[(c0 + n + 1) * 40 + r] = (bf16)f.y;
        Bs[(c0 + n + 2) * 40 + r] = (bf16)f.z;
        Bs[(c0 + n + 3) * 40 + r] = (bf16)f.w;
      }
    }
    __syncthreads();

    v16bf bfr[2];
    #pragma unroll
    for (int nt = 0; nt < 2; ++nt)
      bfr[nt] = load_frag(Bs, wn * 32 + nt * 16, 0, 40, lane);
    #pragma unroll
    for (int mt = 0; mt < 4; ++mt) {
      v16bf afr = load_frag(As, wm * 64 + mt * 16, 0, 40, lane);
      #pragma unroll
      for (int nt = 0; nt < 2; ++nt)
        acc[mt][nt] = wmma_bf16(afr, bfr[nt], acc[mt][nt]);
    }
    __syncthreads();
  }

  // Epilogue: +bias, scatter column n = h*192 + t*64 + d into q/k/v (bf16)
  const int n_lane = lane & 15;
  const int moff   = (lane >> 4) << 3;
  #pragma unroll
  for (int mt = 0; mt < 4; ++mt) {
    #pragma unroll
    for (int nt = 0; nt < 2; ++nt) {
      const int n_g = N0 + wn * 32 + nt * 16 + n_lane;
      const int h = n_g / 192, rem = n_g - h * 192;
      const int t = rem >> 6, d = rem & 63;
      bf16* dst = (t == 0) ? qws : ((t == 1) ? kws : vws);
      const float bias = b_qkv[n_g];
      const int rowbase = M0 + wm * 64 + mt * 16 + moff;
      #pragma unroll
      for (int e = 0; e < 8; ++e) {
        const int g = rowbase + e;
        const int b = g >> 10, s = g & 1023;
        dst[(size_t)((b * NH + h) * S_ + s) * DK + d] = (bf16)(acc[mt][nt][e] + bias);
      }
    }
  }
}

// ---------------------------------------------------------------------------
// Kernel 2: column-softmax stats.  s[i,j] = SCALE * q[i]·k[j];
// m[j] = max_i s ; zws[j] = 1 / sum_i exp(s - m[j]).
// One wave owns 16 columns j; loops i over S in 16-row WMMA tiles.
__global__ __launch_bounds__(256)
void stats_kernel(const bf16* __restrict__ qws, const bf16* __restrict__ kws,
                  float* __restrict__ mws, float* __restrict__ zws)
{
  const int jb = blockIdx.x;           // 0..7
  const int bh = blockIdx.y;           // 0..63
  const int tid = threadIdx.x;
  const int lane = tid & 31, wv = tid >> 5;
  const int j0 = jb * 128 + wv * 16;
  const bf16* qb = qws + (size_t)bh * S_ * DK;
  const bf16* kb = kws + (size_t)bh * S_ * DK;

  // B fragments: k rows j0..j0+15 over d (loaded once)
  const v16bf bk0 = load_frag(kb, j0, 0,  DK, lane);
  const v16bf bk1 = load_frag(kb, j0, 32, DK, lane);
  const v8f zero = {0.f,0.f,0.f,0.f,0.f,0.f,0.f,0.f};

  float runm = -3.0e38f, runZ = 0.f;
  for (int i0 = 0; i0 < S_; i0 += 16) {
    v16bf a0 = load_frag(qb, i0, 0,  DK, lane);
    v16bf a1 = load_frag(qb, i0, 32, DK, lane);
    v8f accv = wmma_bf16(a0, bk0, zero);
    accv = wmma_bf16(a1, bk1, accv);

    float s[8], tmax = -3.0e38f;
    #pragma unroll
    for (int e = 0; e < 8; ++e) { s[e] = accv[e] * ATT_SCALE; tmax = fmaxf(tmax, s[e]); }
    tmax = fmaxf(tmax, __shfl_xor(tmax, 16, 32));  // combine the two M-halves (same column)
    const float newm = fmaxf(runm, tmax);
    runZ *= __expf(runm - newm);
    float part = 0.f;
    #pragma unroll
    for (int e = 0; e < 8; ++e) part += __expf(s[e] - newm);
    part += __shfl_xor(part, 16, 32);
    runZ += part;
    runm = newm;
  }
  if (lane < 16) {
    mws[(size_t)bh * S_ + j0 + lane] = runm;
    zws[(size_t)bh * S_ + j0 + lane] = 1.0f / runZ;
  }
}

// ---------------------------------------------------------------------------
// Kernel 3: res[i,d] = sum_j exp(SCALE*q·k - m[j]) * (1/Z[j]) * v[j,d]
// Block = (b,h, 128-row i block). v^T staged in LDS (64x1024 bf16 = 128KB).
// Each wave owns 16 i rows; P' strip goes through a wave-private LDS buffer.
__global__ __launch_bounds__(256)
void attnout_kernel(const bf16* __restrict__ qws, const bf16* __restrict__ kws,
                    const bf16* __restrict__ vws, const float* __restrict__ mws,
                    const float* __restrict__ zws, bf16* __restrict__ resb)
{
  extern __shared__ bf16 smem[];
  bf16* vT = smem;                    // [64][1024] : vT[d][j] = v[j][d]
  bf16* Pb = smem + (size_t)DK * S_;  // [128][40]  : 8 wave-private 16-row strips

  const int ib = blockIdx.x;          // 0..7
  const int bh = blockIdx.y;          // 0..63
  const int tid = threadIdx.x, lane = tid & 31, wv = tid >> 5;
  const int bIdx = bh >> 3, h = bh & 7;

  const bf16* qb = qws + (size_t)bh * S_ * DK;
  const bf16* kb = kws + (size_t)bh * S_ * DK;
  const bf16* vb = vws + (size_t)bh * S_ * DK;
  const float* mcol = mws + (size_t)bh * S_;
  const float* zcol = zws + (size_t)bh * S_;

  // Stage v^T: each thread transposes 4 rows of v
  #pragma unroll
  for (int rr = 0; rr < 4; ++rr) {
    const int j = tid * 4 + rr;
    const bf16* src = vb + (size_t)j * DK;
    #pragma unroll
    for (int d8 = 0; d8 < DK; d8 += 8) {
      v8bf vvec = *(const v8bf*)(src + d8);
      #pragma unroll
      for (int e = 0; e < 8; ++e)
        vT[(size_t)(d8 + e) * S_ + j] = vvec[e];
    }
  }
  __syncthreads();

  const int i0 = ib * 128 + wv * 16;
  const v16bf aq0 = load_frag(qb, i0, 0,  DK, lane);
  const v16bf aq1 = load_frag(qb, i0, 32, DK, lane);
  const v8f zero = {0.f,0.f,0.f,0.f,0.f,0.f,0.f,0.f};

  v8f o[4];
  #pragma unroll
  for (int nt = 0; nt < 4; ++nt) o[nt] = zero;

  bf16* Pw = Pb + wv * 16 * 40;       // wave-private strip: 16 x 32 (stride 40)
  const int n_lane = lane & 15;
  const int moff   = (lane >> 4) << 3;

  for (int j0 = 0; j0 < S_; j0 += 32) {
    // s strip 16(i) x 32(j), two 16x16 WMMA tiles; exp + scale; spill bf16 P'
    #pragma unroll
    for (int js = 0; js < 32; js += 16) {
      v16bf bkk0 = load_frag(kb, j0 + js, 0,  DK, lane);
      v16bf bkk1 = load_frag(kb, j0 + js, 32, DK, lane);
      v8f sa = wmma_bf16(aq0, bkk0, zero);
      sa = wmma_bf16(aq1, bkk1, sa);
      const int jc = j0 + js + n_lane;
      const float mc = mcol[jc];
      const float rz = zcol[jc];
      #pragma unroll
      for (int e = 0; e < 8; ++e) {
        const float p = __expf(sa[e] * ATT_SCALE - mc) * rz;
        Pw[(moff + e) * 40 + js + n_lane] = (bf16)p;
      }
    }
    // Reload P' as A fragment (wave-local LDS: no workgroup barrier needed)
    v16bf ap = load_frag(Pw, 0, 0, 40, lane);
    #pragma unroll
    for (int nt = 0; nt < 4; ++nt) {
      v16bf bv = load_frag(vT, nt * 16, j0, S_, lane);
      o[nt] = wmma_bf16(ap, bv, o[nt]);
    }
  }

  // Write 16x64 result to resb[B*S, 512] (row = b*S+i, col = h*64+d), bf16
  #pragma unroll
  for (int nt = 0; nt < 4; ++nt) {
    const int dcol = nt * 16 + n_lane;
    #pragma unroll
    for (int e = 0; e < 8; ++e) {
      const int i = i0 + moff + e;
      resb[(size_t)(bIdx * S_ + i) * INNER + h * DK + dcol] = (bf16)o[nt][e];
    }
  }
}

// ---------------------------------------------------------------------------
// Kernel 4: out = resb @ w_out + b_out + xt, written transposed as [B,C,H,W].
// A fragments read straight from global (resb is bf16 row-major over K).
__global__ __launch_bounds__(256)
void proj_kernel(const bf16* __restrict__ resb, const float* __restrict__ w_out,
                 const float* __restrict__ b_out, const float* __restrict__ x,
                 float* __restrict__ out)
{
  const int M0 = blockIdx.x * 128;
  const int N0 = blockIdx.y * 128;
  const int tid = threadIdx.x, lane = tid & 31, wv = tid >> 5;
  const int wm = wv & 1, wn = wv >> 1;

  __shared__ bf16 Bs[128 * 40];       // w_out^T tile [n][k]

  v8f acc[4][2];
  #pragma unroll
  for (int i = 0; i < 4; ++i)
    #pragma unroll
    for (int j = 0; j < 2; ++j)
      acc[i][j] = v8f{0.f,0.f,0.f,0.f,0.f,0.f,0.f,0.f};

  for (int k0 = 0; k0 < INNER; k0 += 32) {
    {
      const int r  = tid >> 3;
      const int c0 = (tid & 7) << 4;
      const float* src = w_out + ((size_t)(k0 + r) * C_ + N0 + c0);
      if (k0 + 32 < INNER) __builtin_prefetch(src + 32 * C_, 0, 1);
      #pragma unroll
      for (int n = 0; n < 16; n += 4) {
        float4 f = *(const float4*)(src + n);
        Bs[(c0 + n + 0) * 40 + r] = (bf16)f.x;
        Bs[(c0 + n + 1) * 40 + r] = (bf16)f.y;
        Bs[(c0 + n + 2) * 40 + r] = (bf16)f.z;
        Bs[(c0 + n + 3) * 40 + r] = (bf16)f.w;
      }
    }
    __syncthreads();

    v16bf bfr[2];
    #pragma unroll
    for (int nt = 0; nt < 2; ++nt)
      bfr[nt] = load_frag(Bs, wn * 32 + nt * 16, 0, 40, lane);
    #pragma unroll
    for (int mt = 0; mt < 4; ++mt) {
      v16bf afr = load_frag(resb, M0 + wm * 64 + mt * 16, k0, INNER, lane);
      #pragma unroll
      for (int nt = 0; nt < 2; ++nt)
        acc[mt][nt] = wmma_bf16(afr, bfr[nt], acc[mt][nt]);
    }
    __syncthreads();
  }

  // Epilogue: + bias + residual x (already in [B,C,S] layout), store transposed
  const int n_lane = lane & 15;
  const int moff   = (lane >> 4) << 3;
  #pragma unroll
  for (int mt = 0; mt < 4; ++mt) {
    #pragma unroll
    for (int nt = 0; nt < 2; ++nt) {
      const int n_g = N0 + wn * 32 + nt * 16 + n_lane;
      const float bias = b_out[n_g];
      const int rowbase = M0 + wm * 64 + mt * 16 + moff;
      #pragma unroll
      for (int e = 0; e < 8; ++e) {
        const int g = rowbase + e;
        const int b = g >> 10, s = g & 1023;
        const size_t idx = (size_t)(b * C_ + n_g) * S_ + s;
        out[idx] = acc[mt][nt][e] + bias + x[idx];
      }
    }
  }
}

// ---------------------------------------------------------------------------
extern "C" void kernel_launch(void* const* d_in, const int* in_sizes, int n_in,
                              void* d_out, int out_size, void* d_ws, size_t ws_size,
                              hipStream_t stream)
{
  (void)in_sizes; (void)n_in; (void)out_size; (void)ws_size;
  const float* x     = (const float*)d_in[0];
  const float* w_qkv = (const float*)d_in[1];
  const float* b_qkv = (const float*)d_in[2];
  const float* w_out = (const float*)d_in[3];
  const float* b_out = (const float*)d_in[4];
  float* out = (float*)d_out;

  // Workspace carve (~34 MB total): q/k/v [B,H,S,DK] bf16, stats, res bf16
  char* p = (char*)d_ws;
  auto carve = [&](size_t bytes) -> char* {
    char* r = p; p += (bytes + 255) & ~(size_t)255; return r;
  };
  const size_t headElems = (size_t)B_ * NH * S_ * DK;       // 4,194,304
  bf16*  qws  = (bf16*)carve(headElems * 2);
  bf16*  kws  = (bf16*)carve(headElems * 2);
  bf16*  vws  = (bf16*)carve(headElems * 2);
  float* mws  = (float*)carve((size_t)B_ * NH * S_ * 4);
  float* zws  = (float*)carve((size_t)B_ * NH * S_ * 4);
  bf16*  resb = (bf16*)carve((size_t)B_ * S_ * INNER * 2);

  qkv_kernel<<<dim3(64, 12), 256, 0, stream>>>(x, w_qkv, b_qkv, qws, kws, vws);
  stats_kernel<<<dim3(8, 64), 256, 0, stream>>>(qws, kws, mws, zws);
  const size_t smB2 = ((size_t)DK * S_ + 128 * 40) * sizeof(bf16);  // ~138 KB
  attnout_kernel<<<dim3(8, 64), 256, smB2, stream>>>(qws, kws, vws, mws, zws, resb);
  proj_kernel<<<dim3(64, 2), 256, 0, stream>>>(resb, w_out, b_out, x, out);
}
